// MOELayer_53772990545994
// MI455X (gfx1250) — matrix operand order repaired and checked by reading, not compiled, not run on hardware
//
#include <hip/hip_runtime.h>

// ---------------- problem constants (from reference setup_inputs) ------------
#define S_TOK 4096   // B*T
#define MDIM  1024   // model dim
#define NEXP  8      // experts
#define FDIM  4096   // ffn dim
#define CAP   1024   // 2*S/E

// Use CDNA5 GLOBAL_LOAD_ASYNC_TO_LDS_B128 (ASYNCcnt) for the A-slab staging.
#define USE_ASYNC_LDS 1

typedef __bf16 bf16;
typedef __attribute__((ext_vector_type(16))) __bf16 v16bf;
typedef __attribute__((ext_vector_type(8)))  __bf16 v8bf;
typedef __attribute__((ext_vector_type(4)))  __bf16 v4bf;
typedef __attribute__((ext_vector_type(2)))  __bf16 v2bf;
typedef __attribute__((ext_vector_type(8)))  float  v8f;

#if USE_ASYNC_LDS
// generic pointer -> 32-bit LDS offset (addrspacecast to AS3, then truncate)
__device__ __forceinline__ unsigned lds_off(const void* p) {
  return (unsigned)(__SIZE_TYPE__)(const __attribute__((address_space(3))) void*)p;
}
#endif

// ============================================================================
// Kernel 1: gating — logits = x @ wg, softmax, top-1/top-2 indices.
// One wave (32 lanes) per token; 8 waves per block.
// ============================================================================
__global__ __launch_bounds__(256) void moe_gate(const float* __restrict__ x,
                                                const float* __restrict__ wg,
                                                float* __restrict__ gates,
                                                int* __restrict__ idx1,
                                                int* __restrict__ idx2) {
  const int lane = threadIdx.x & 31;
  const int wave = threadIdx.x >> 5;
  const int s = blockIdx.x * 8 + wave;
  const float* xr = x + (size_t)s * MDIM;

  float acc[NEXP];
#pragma unroll
  for (int e = 0; e < NEXP; ++e) acc[e] = 0.f;

  for (int m = lane; m < MDIM; m += 32) {
    float xv = xr[m];
    const float4* wp = (const float4*)(wg + (size_t)m * NEXP);
    float4 w0 = wp[0];
    float4 w1v = wp[1];
    acc[0] += xv * w0.x;  acc[1] += xv * w0.y;
    acc[2] += xv * w0.z;  acc[3] += xv * w0.w;
    acc[4] += xv * w1v.x; acc[5] += xv * w1v.y;
    acc[6] += xv * w1v.z; acc[7] += xv * w1v.w;
  }
#pragma unroll
  for (int e = 0; e < NEXP; ++e) {
#pragma unroll
    for (int off = 16; off > 0; off >>= 1) acc[e] += __shfl_xor(acc[e], off, 32);
  }

  if (lane == 0) {
    float mx = acc[0];
#pragma unroll
    for (int e = 1; e < NEXP; ++e) mx = fmaxf(mx, acc[e]);
    float gg[NEXP];
    float sum = 0.f;
#pragma unroll
    for (int e = 0; e < NEXP; ++e) { gg[e] = __expf(acc[e] - mx); sum += gg[e]; }
    float inv = 1.f / sum;
#pragma unroll
    for (int e = 0; e < NEXP; ++e) gates[(size_t)s * NEXP + e] = gg[e] * inv;
    // argmax (first max wins, like jnp.argmax)
    int a1 = 0; float b1v = acc[0];
#pragma unroll
    for (int e = 1; e < NEXP; ++e) if (acc[e] > b1v) { b1v = acc[e]; a1 = e; }
    int a2 = -1; float b2v = -3.4e38f;
#pragma unroll
    for (int e = 0; e < NEXP; ++e) if (e != a1 && acc[e] > b2v) { b2v = acc[e]; a2 = e; }
    idx1[s] = a1;
    idx2[s] = a2;
  }
}

// ============================================================================
// Kernel 2: sequential slot assignment (token-order cumsum), capacity drop,
// gate renormalization, and aux loss. Single wave; lane e keeps expert e's
// counter, __shfl broadcasts the slot to the token owner.
// ============================================================================
__global__ __launch_bounds__(32) void moe_assign(const float* __restrict__ gates,
                                                 const int* __restrict__ idx1,
                                                 const int* __restrict__ idx2,
                                                 int* __restrict__ slot1,
                                                 int* __restrict__ slot2,
                                                 float* __restrict__ g1o,
                                                 float* __restrict__ g2o,
                                                 float* __restrict__ laux_out) {
  const int lane = threadIdx.x;

  // me[e] partial sums: lane l accumulates elements i == l (mod 32); expert = i%8
  float gs = 0.f;
  for (int i = lane; i < S_TOK * NEXP; i += 32) gs += gates[i];
  gs += __shfl_xor(gs, 16, 32);
  gs += __shfl_xor(gs, 8, 32);   // lanes 0..7 now hold per-expert sums

  // ---- pass 1: loc1 = cumsum(mask1)-1 (pre-capacity), store raw ----
  int cnt1 = 0;
  for (int s0 = 0; s0 < S_TOK; s0 += 32) {
    int vi1 = idx1[s0 + lane];
    int myloc = 0;
#pragma unroll 1
    for (int t = 0; t < 32; ++t) {
      int i1 = __shfl(vi1, t, 32);
      int loc = __shfl(cnt1, i1, 32);
      if (lane == i1) cnt1++;
      if (lane == t) myloc = loc;
    }
    slot1[s0 + lane] = myloc;   // raw loc1 (same thread re-reads it in pass 2)
  }

  // aux loss: l_aux = E * sum_e (me[e] * ce[e]),  me = gsum/S, ce = cnt1/S
  float part = (lane < NEXP) ? (gs / (float)S_TOK) * ((float)cnt1 / (float)S_TOK) : 0.f;
  part += __shfl_xor(part, 1, 32);
  part += __shfl_xor(part, 2, 32);
  part += __shfl_xor(part, 4, 32);
  if (lane == 0) laux_out[0] = part * (float)NEXP;

  // ---- pass 2: loc2 = cumsum(mask2)-1 + sum(mask1), finalize ----
  const int base = cnt1;   // per-expert total of mask1 (pre-drop)
  int c2 = 0;
  for (int s0 = 0; s0 < S_TOK; s0 += 32) {
    int vi1 = idx1[s0 + lane];
    int vi2 = idx2[s0 + lane];
    int myloc2 = 0;
#pragma unroll 1
    for (int t = 0; t < 32; ++t) {
      int i2 = __shfl(vi2, t, 32);
      int cur = c2 + base;
      int loc = __shfl(cur, i2, 32);
      if (lane == i2) c2++;
      if (lane == t) myloc2 = loc;
    }
    const int s = s0 + lane;
    int l1 = slot1[s];
    int l2 = myloc2;
    bool k1 = (l1 < CAP);
    bool k2 = (l2 < CAP);
    float ga = gates[(size_t)s * NEXP + vi1];
    float gb = gates[(size_t)s * NEXP + vi2];
    float g1s = k1 ? ga : 0.f;
    float g2s = k2 ? gb : 0.f;
    float den = g1s + g2s;
    den = (den < 1e-9f) ? 1e-9f : den;
    slot1[s] = k1 ? l1 : -1;
    slot2[s] = k2 ? l2 : -1;
    g1o[s] = g1s / den;
    g2o[s] = g2s / den;
  }
}

// ============================================================================
// Kernel 3: fp32 -> bf16 (RNE) weight conversion, 4 elements/thread.
// ============================================================================
__global__ __launch_bounds__(256) void cvt_f32_bf16(const float* __restrict__ src,
                                                    unsigned short* __restrict__ dst,
                                                    int n4) {
  int i = blockIdx.x * 256 + threadIdx.x;
  if (i >= n4) return;
  float4 v = ((const float4*)src)[i];
  v4bf o = { (bf16)v.x, (bf16)v.y, (bf16)v.z, (bf16)v.w };
  ((v4bf*)dst)[i] = o;
}

// Kernel 4: zero workspace region (uint4 granularity)
__global__ __launch_bounds__(256) void zero_ws(uint4* __restrict__ p, int n16) {
  int i = blockIdx.x * 256 + threadIdx.x;
  if (i < n16) p[i] = make_uint4(0u, 0u, 0u, 0u);
}

// ============================================================================
// Kernel 5: dispatch tokens into [E, C, M] bf16 expert buffers.
// One block per token, 4 elements per thread.
// ============================================================================
__global__ __launch_bounds__(256) void moe_dispatch(const float* __restrict__ x,
                                                    const int* __restrict__ idx1,
                                                    const int* __restrict__ idx2,
                                                    const int* __restrict__ slot1,
                                                    const int* __restrict__ slot2,
                                                    unsigned short* __restrict__ xdisp) {
  const int s = blockIdx.x;
  const int t = threadIdx.x;
  float4 v = ((const float4*)(x + (size_t)s * MDIM))[t];
  v4bf b = { (bf16)v.x, (bf16)v.y, (bf16)v.z, (bf16)v.w };
  int sl1 = slot1[s];
  if (sl1 >= 0) {
    int e1 = idx1[s];
    ((v4bf*)(xdisp + ((size_t)e1 * CAP + sl1) * MDIM))[t] = b;
  }
  int sl2 = slot2[s];
  if (sl2 >= 0) {
    int e2 = idx2[s];
    ((v4bf*)(xdisp + ((size_t)e2 * CAP + sl2) * MDIM))[t] = b;
  }
}

// ============================================================================
// Kernel 6: batched per-expert GEMM, v_wmma_f32_16x16x32_bf16.
//   A: [E][1024][KDIM] bf16 row-major, B: [E][KDIM][NDIM] bf16 row-major.
//   Block tile 128x256, K-slab 32, double-buffered LDS, 8 waves; each wave
//   computes 64x64 (4x4 accumulators -> 16 WMMA per k-step).
//   A slab staged with GLOBAL_LOAD_ASYNC_TO_LDS_B128 (ASYNCcnt) when enabled;
//   B slab staged via registers with on-the-fly [k][n]->[n][k] transpose so B
//   fragments are contiguous 32B/lane (k = 16*half + s, same order as A).
//   LDS row stride 40 bf16 (80B) keeps 16B alignment and spreads banks.
// ============================================================================
template <int KDIM, int NDIM, bool RELU, bool OUTBF16>
__global__ __launch_bounds__(256) void moe_gemm(const unsigned short* __restrict__ Au,
                                                const unsigned short* __restrict__ Bu,
                                                unsigned short* __restrict__ Obf,
                                                float* __restrict__ Of32) {
  constexpr int BM = 128, BN = 256, BK = 32, LSTR = 40;
  __shared__ __align__(16) bf16 Alds[2][BM * LSTR];   // 2 x 10 KB
  __shared__ __align__(16) bf16 Blds[2][BN * LSTR];   // 2 x 20 KB

  const int e = blockIdx.z;
  const int mtile = blockIdx.y * BM;
  const int ntile = blockIdx.x * BN;
  const bf16* A = (const bf16*)Au + (size_t)e * 1024 * KDIM;
  const bf16* B = (const bf16*)Bu + (size_t)e * KDIM * NDIM;

  const int tid = threadIdx.x;
  const int lane = tid & 31;
  const int wave = tid >> 5;
  const int wm = (wave & 1) * 64;   // wave M offset in block tile
  const int wn = (wave >> 1) * 64;  // wave N offset in block tile

  v8f acc[4][4] = {};

  // staging roles
  const int arow = tid >> 1;          // 0..127
  const int achk = (tid & 1) * 16;    // 0 / 16 within K-slab
  const int bk = (tid >> 4) * 2;      // 0,2,..,30 (pair of K rows)
  const int bn = (tid & 15) * 16;     // 0..240

  const int fr = lane & 15;           // fragment row/col within 16
  const int kq = lane >> 4;           // K-half selector: k = 16*kq + s (contiguous order)

  const bf16* agp = A + (size_t)(mtile + arow) * KDIM + achk;
  const bf16* bgp = B + (size_t)bk * NDIM + ntile + bn;

  // ---- prologue: fetch slab 0 ----
#if USE_ASYNC_LDS
  {
    unsigned l0 = lds_off(&Alds[0][arow * LSTR + achk]);
    asm volatile("global_load_async_to_lds_b128 %0, %1, off"
                 :: "v"(l0), "v"(agp) : "memory");
    asm volatile("global_load_async_to_lds_b128 %0, %1, off offset:16"
                 :: "v"(l0), "v"(agp) : "memory");
  }
#else
  v8bf pa0 = *(const v8bf*)agp;
  v8bf pa1 = *(const v8bf*)(agp + 8);
#endif
  v8bf pb00 = *(const v8bf*)bgp;
  v8bf pb01 = *(const v8bf*)(bgp + 8);
  v8bf pb10 = *(const v8bf*)(bgp + NDIM);
  v8bf pb11 = *(const v8bf*)(bgp + NDIM + 8);

  int cur = 0;
  for (int k0 = 0; k0 < KDIM; k0 += BK, cur ^= 1) {
    const bool has_next = (k0 + BK) < KDIM;

    // ---- commit staged B slab (registers -> LDS[cur], transposed) ----
#pragma unroll
    for (int q = 0; q < 8; ++q) {
      v2bf p0 = { pb00[q], pb10[q] };
      *(v2bf*)&Blds[cur][(bn + q) * LSTR + bk] = p0;
      v2bf p1 = { pb01[q], pb11[q] };
      *(v2bf*)&Blds[cur][(bn + 8 + q) * LSTR + bk] = p1;
    }
#if !USE_ASYNC_LDS
    *(v8bf*)&Alds[cur][arow * LSTR + achk] = pa0;
    *(v8bf*)&Alds[cur][arow * LSTR + achk + 8] = pa1;
#endif

    // ---- prefetch next B slab into registers (overlaps with compute) ----
    if (has_next) {
      bgp += (size_t)BK * NDIM;
      pb00 = *(const v8bf*)bgp;
      pb01 = *(const v8bf*)(bgp + 8);
      pb10 = *(const v8bf*)(bgp + NDIM);
      pb11 = *(const v8bf*)(bgp + NDIM + 8);
#if !USE_ASYNC_LDS
      agp += BK;
      pa0 = *(const v8bf*)agp;
      pa1 = *(const v8bf*)(agp + 8);
#endif
    }

#if USE_ASYNC_LDS
    // this slab's async A data must have landed before any thread reads it
    asm volatile("s_wait_asynccnt 0x0" ::: "memory");
#endif
    __syncthreads();

#if USE_ASYNC_LDS
    // issue next A slab into the other buffer (safe: after barrier, the other
    // buffer's readers from the previous iteration are all done)
    if (has_next) {
      const bf16* an = agp + BK;
      unsigned l1 = lds_off(&Alds[cur ^ 1][arow * LSTR + achk]);
      asm volatile("global_load_async_to_lds_b128 %0, %1, off"
                   :: "v"(l1), "v"(an) : "memory");
      asm volatile("global_load_async_to_lds_b128 %0, %1, off offset:16"
                   :: "v"(l1), "v"(an) : "memory");
      agp = an;
    }
#endif

    // ---- compute from LDS[cur]: 4 B frags + 4 A frags -> 16 WMMA ----
    v16bf bfr[4];
#pragma unroll
    for (int j = 0; j < 4; ++j) {
      const bf16* bp = &Blds[cur][(wn + j * 16 + fr) * LSTR + kq * 16];
      v8bf blo = *(const v8bf*)bp;
      v8bf bhi = *(const v8bf*)(bp + 8);
#pragma unroll
      for (int q = 0; q < 8; ++q) { bfr[j][q] = blo[q]; bfr[j][q + 8] = bhi[q]; }
    }
#pragma unroll
    for (int i = 0; i < 4; ++i) {
      const bf16* ap = &Alds[cur][(wm + i * 16 + fr) * LSTR + kq * 16];
      v8bf lo = *(const v8bf*)ap;         // K 0-7  / 16-23
      v8bf hi = *(const v8bf*)(ap + 8);   // K 8-15 / 24-31
      v16bf af;
#pragma unroll
      for (int q = 0; q < 8; ++q) { af[q] = lo[q]; af[q + 8] = hi[q]; }
#pragma unroll
      for (int j = 0; j < 4; ++j) {
        acc[i][j] = __builtin_amdgcn_wmma_f32_16x16x32_bf16(
            /*neg_a=*/false, af, /*neg_b=*/false, bfr[j],
            /*c_mod=*/(short)0, acc[i][j],
            /*reuse_a=*/false, /*reuse_b=*/false);
      }
    }
    __syncthreads();
  }

  // ---- epilogue: C/D layout -> VGPR v holds row v (lanes 0-15) / v+8 (16-31)
  const int half = lane >> 4;
#pragma unroll
  for (int i = 0; i < 4; ++i) {
#pragma unroll
    for (int j = 0; j < 4; ++j) {
      const int mb = mtile + wm + i * 16 + half * 8;
      const int nb = ntile + wn + j * 16 + fr;
#pragma unroll
      for (int v = 0; v < 8; ++v) {
        float val = acc[i][j][v];
        if (RELU) val = fmaxf(val, 0.f);
        const size_t off = (size_t)e * 1024 * NDIM + (size_t)(mb + v) * NDIM + nb;
        if (OUTBF16) ((bf16*)Obf)[off] = (bf16)val;
        else         Of32[off] = val;
      }
    }
  }
}

// ============================================================================
// Kernel 7: combine — out[s,:] = g1*eout[e1][slot1] + g2*eout[e2][slot2]
// ============================================================================
__global__ __launch_bounds__(256) void moe_combine(const float* __restrict__ eout,
                                                   const int* __restrict__ idx1,
                                                   const int* __restrict__ idx2,
                                                   const int* __restrict__ slot1,
                                                   const int* __restrict__ slot2,
                                                   const float* __restrict__ g1,
                                                   const float* __restrict__ g2,
                                                   float* __restrict__ out) {
  const int s = blockIdx.x;
  const int t = threadIdx.x;
  float4 r = make_float4(0.f, 0.f, 0.f, 0.f);
  int sl1 = slot1[s];
  if (sl1 >= 0) {
    float cw = g1[s];
    float4 a = ((const float4*)(eout + ((size_t)idx1[s] * CAP + sl1) * MDIM))[t];
    r.x += cw * a.x; r.y += cw * a.y; r.z += cw * a.z; r.w += cw * a.w;
  }
  int sl2 = slot2[s];
  if (sl2 >= 0) {
    float cw = g2[s];
    float4 a = ((const float4*)(eout + ((size_t)idx2[s] * CAP + sl2) * MDIM))[t];
    r.x += cw * a.x; r.y += cw * a.y; r.z += cw * a.z; r.w += cw * a.w;
  }
  ((float4*)(out + (size_t)s * MDIM))[t] = r;
}

// ============================================================================
// host launch
// ============================================================================
extern "C" void kernel_launch(void* const* d_in, const int* in_sizes, int n_in,
                              void* d_out, int out_size, void* d_ws, size_t ws_size,
                              hipStream_t stream) {
  (void)in_sizes; (void)n_in; (void)out_size; (void)ws_size;
  const float* x  = (const float*)d_in[0];  // [4,1024,1024]
  const float* wg = (const float*)d_in[1];  // [1024,8]
  const float* w1 = (const float*)d_in[2];  // [8,1024,4096]
  const float* w2 = (const float*)d_in[3];  // [8,4096,1024]
  float* out = (float*)d_out;               // [S*M] combined + [1] l_aux

  char* ws = (char*)d_ws;
  size_t off = 0;
  auto take = [&](size_t bytes) -> char* {
    char* p = ws + off;
    off = (off + bytes + 255) & ~(size_t)255;
    return p;
  };

  float* gates = (float*)take((size_t)S_TOK * NEXP * 4);
  int* idx1    = (int*)take((size_t)S_TOK * 4);
  int* idx2    = (int*)take((size_t)S_TOK * 4);
  int* slot1   = (int*)take((size_t)S_TOK * 4);
  int* slot2   = (int*)take((size_t)S_TOK * 4);
  float* g1f   = (float*)take((size_t)S_TOK * 4);
  float* g2f   = (float*)take((size_t)S_TOK * 4);
  unsigned short* w1b   = (unsigned short*)take((size_t)NEXP * MDIM * FDIM * 2);
  unsigned short* w2b   = (unsigned short*)take((size_t)NEXP * FDIM * MDIM * 2);
  unsigned short* xdisp = (unsigned short*)take((size_t)NEXP * CAP * MDIM * 2);
  unsigned short* hbuf  = (unsigned short*)take((size_t)NEXP * CAP * FDIM * 2);
  float* eout  = (float*)take((size_t)NEXP * CAP * MDIM * 4);

  // 1) gating
  moe_gate<<<S_TOK / 8, 256, 0, stream>>>(x, wg, gates, idx1, idx2);
  // 2) sequential slot assignment + l_aux (written at out[S*M])
  moe_assign<<<1, 32, 0, stream>>>(gates, idx1, idx2, slot1, slot2, g1f, g2f,
                                   out + (size_t)S_TOK * MDIM);
  // 3) weight conversion to bf16
  {
    const int n4 = NEXP * MDIM * FDIM / 4;  // 8,388,608
    cvt_f32_bf16<<<(n4 + 255) / 256, 256, 0, stream>>>(w1, w1b, n4);
    cvt_f32_bf16<<<(n4 + 255) / 256, 256, 0, stream>>>(w2, w2b, n4);
  }
  // 4) zero dispatch buffers, then scatter tokens
  {
    const int n16 = (int)((size_t)NEXP * CAP * MDIM * 2 / 16);  // 1,048,576
    zero_ws<<<(n16 + 255) / 256, 256, 0, stream>>>((uint4*)xdisp, n16);
  }
  moe_dispatch<<<S_TOK, 256, 0, stream>>>(x, idx1, idx2, slot1, slot2, xdisp);
  // 5) expert FFN: h = relu(xdisp @ w1) ; eout = h @ w2
  moe_gemm<MDIM, FDIM, true, true><<<dim3(FDIM / 256, CAP / 128, NEXP), 256, 0, stream>>>(
      xdisp, w1b, hbuf, (float*)nullptr);
  moe_gemm<FDIM, MDIM, false, false><<<dim3(MDIM / 256, CAP / 128, NEXP), 256, 0, stream>>>(
      hbuf, w2b, (unsigned short*)nullptr, eout);
  // 6) combine back to [S, M]
  moe_combine<<<S_TOK, 256, 0, stream>>>(eout, idx1, idx2, slot1, slot2, g1f, g2f, out);
}
